// TextureWarpingModule_78408922956452
// MI455X (gfx1250) — matrix operand order repaired and checked by reading, not compiled, not run on hardware
//
#include <hip/hip_runtime.h>
#include <hip/hip_bf16.h>

#define HW   4096
#define NPIX 8192
#define CHN  256

typedef __attribute__((ext_vector_type(16))) _Float16 v16h;
typedef __attribute__((ext_vector_type(8)))  _Float16 v8h;
typedef __attribute__((ext_vector_type(8)))  float    v8f;

union V16H { v16h v; v8h h[2]; _Float16 e[16]; };
union V8F  { v8f v; float e[8]; };

// ---------------------------------------------------------------------------
// Fragment loaders (wave32; A/B use this code's contiguous k order, C/D per CDNA5 ISA §7.12.2)
// A 16x32 f16: lane m=lane&15; elems j -> K[half*16 + j]
// B 32x16 f16: lane n=lane&15; elems j -> K[half*16 + j]
// C/D 16x16 f32: vgpr r -> M = half*8 + r, N = lane&15
// ---------------------------------------------------------------------------
__device__ __forceinline__ v16h load_frag_pair(const _Float16* p0, const _Float16* p1) {
  V16H u;
  u.h[0] = *(const v8h*)p0;
  u.h[1] = *(const v8h*)p1;
  return u.v;
}

// ---------------------------------------------------------------------------
// f16 NT GEMM: out[M,N] = A[M,K] * B[N,K]^T + bias, N = NPIX.
// Register-blocked: each wave computes a 32(M) x 64(N) tile = 2x4 WMMA tiles
// (8 independent accumulator chains, 24B operand traffic per WMMA).
// grid: (NPIX/512, Mp/32), block 256 (8 waves).
// ---------------------------------------------------------------------------
__global__ __launch_bounds__(256)
void gemm_nt_wmma(const _Float16* __restrict__ A,
                  const _Float16* __restrict__ Bm,
                  const float* __restrict__ bias,
                  float* __restrict__ outf,
                  _Float16* __restrict__ outh, int Cs, int coff,
                  int Mout, int K) {
  const int lane = threadIdx.x & 31;
  const int wv   = threadIdx.x >> 5;
  const int hf   = lane >> 4;
  const int mr   = lane & 15;
  const int nbase = (blockIdx.x * 8 + wv) * 64;
  const int mbase = blockIdx.y * 32;

  const _Float16* Ar0 = A + (size_t)(mbase + mr) * K;
  const _Float16* Ar1 = A + (size_t)(mbase + 16 + mr) * K;
  const _Float16* Br[4];
#pragma unroll
  for (int j = 0; j < 4; ++j)
    Br[j] = Bm + (size_t)(nbase + j * 16 + mr) * K + hf * 16;

  V8F acc[2][4];
#pragma unroll
  for (int i = 0; i < 2; ++i)
#pragma unroll
    for (int r = 0; r < 8; ++r) {
      int m = mbase + i * 16 + hf * 8 + r;
      float bv = (m < Mout) ? bias[m] : 0.0f;
#pragma unroll
      for (int j = 0; j < 4; ++j) acc[i][j].e[r] = bv;
    }

  for (int kt = 0; kt < K; kt += 32) {
    v16h a0 = load_frag_pair(Ar0 + kt + hf * 16, Ar0 + kt + hf * 16 + 8);
    v16h a1 = load_frag_pair(Ar1 + kt + hf * 16, Ar1 + kt + hf * 16 + 8);
#pragma unroll
    for (int j = 0; j < 4; ++j) {
      v16h b = load_frag_pair(Br[j] + kt, Br[j] + kt + 8);
      acc[0][j].v = __builtin_amdgcn_wmma_f32_16x16x32_f16(false, a0, false, b,
                                                           (short)0, acc[0][j].v, false, false);
      acc[1][j].v = __builtin_amdgcn_wmma_f32_16x16x32_f16(false, a1, false, b,
                                                           (short)0, acc[1][j].v, false, false);
    }
  }
  const int bidx = nbase >> 12;
  const int hwb  = nbase & (HW - 1);
#pragma unroll
  for (int i = 0; i < 2; ++i)
#pragma unroll
    for (int r = 0; r < 8; ++r) {
      int m = mbase + i * 16 + hf * 8 + r;
      if (m < Mout) {
#pragma unroll
        for (int j = 0; j < 4; ++j) {
          int hw = hwb + j * 16 + mr;
          float vv = acc[i][j].e[r];
          if (outf) outf[((size_t)bidx * Mout + m) * HW + hw] = vv;
          if (outh) outh[(size_t)(bidx * HW + hw) * Cs + coff + m] = (_Float16)vv;
        }
      }
    }
}

// ---------------------------------------------------------------------------
// Implicit-im2col 3x3 conv (pad 1) as WMMA GEMM, 32(M) x 64(N) per wave.
// All 64 N-columns of a wave share one image row -> y-bounds check is
// wave-uniform (skip whole tap); x-bounds stays per-lane (zero fragment).
// Wp: [Mp][9][C] f16 (zero rows for m>=Mout), Xh: [NPIX][C] f16 NHWC.
// grid: (NPIX/512, Mp/32), block 256.
// ---------------------------------------------------------------------------
__global__ __launch_bounds__(256)
void conv3x3_wmma(const _Float16* __restrict__ Wp,
                  const _Float16* __restrict__ Xh,
                  const float* __restrict__ bias,
                  float* __restrict__ outf,
                  _Float16* __restrict__ outh, int Cs, int coff,
                  int Mout, int C) {
  const int lane = threadIdx.x & 31;
  const int wv   = threadIdx.x >> 5;
  const int hf   = lane >> 4;
  const int mr   = lane & 15;
  const int nbase = (blockIdx.x * 8 + wv) * 64;
  const int mbase = blockIdx.y * 32;
  const int bp = nbase >> 12;
  const int y  = (nbase & 4095) >> 6;

  V8F acc[2][4];
#pragma unroll
  for (int i = 0; i < 2; ++i)
#pragma unroll
    for (int r = 0; r < 8; ++r) {
      int m = mbase + i * 16 + hf * 8 + r;
      float bv = (m < Mout) ? bias[m] : 0.0f;
#pragma unroll
      for (int j = 0; j < 4; ++j) acc[i][j].e[r] = bv;
    }

#pragma unroll
  for (int kp = 0; kp < 9; ++kp) {
    const int dy = kp / 3 - 1, dx = kp % 3 - 1;
    const int yy = y + dy;
    if (yy < 0 || yy >= 64) continue;   // wave-uniform: skip whole tap
    const _Float16* rowb = Xh + ((size_t)bp * HW + yy * 64) * C + hf * 16;
    const _Float16* Aw0 = Wp + ((size_t)(mbase + mr) * 9 + kp) * C;
    const _Float16* Aw1 = Wp + ((size_t)(mbase + 16 + mr) * 9 + kp) * C;
    int  xx[4];
    bool ok[4];
#pragma unroll
    for (int j = 0; j < 4; ++j) {
      xx[j] = j * 16 + mr + dx;
      ok[j] = (xx[j] >= 0) && (xx[j] < 64);
    }
    for (int kt = 0; kt < C; kt += 32) {
      v16h a0 = load_frag_pair(Aw0 + kt + hf * 16, Aw0 + kt + hf * 16 + 8);
      v16h a1 = load_frag_pair(Aw1 + kt + hf * 16, Aw1 + kt + hf * 16 + 8);
#pragma unroll
      for (int j = 0; j < 4; ++j) {
        V16H bu;
#pragma unroll
        for (int t = 0; t < 16; ++t) bu.e[t] = (_Float16)0.0f;
        if (ok[j]) {
          const _Float16* bp2 = rowb + (size_t)xx[j] * C + kt;
          bu.h[0] = *(const v8h*)(bp2);
          bu.h[1] = *(const v8h*)(bp2 + 8);
        }
        acc[0][j].v = __builtin_amdgcn_wmma_f32_16x16x32_f16(false, a0, false, bu.v,
                                                             (short)0, acc[0][j].v, false, false);
        acc[1][j].v = __builtin_amdgcn_wmma_f32_16x16x32_f16(false, a1, false, bu.v,
                                                             (short)0, acc[1][j].v, false, false);
      }
    }
  }
  const int hwb = nbase & (HW - 1);
#pragma unroll
  for (int i = 0; i < 2; ++i)
#pragma unroll
    for (int r = 0; r < 8; ++r) {
      int m = mbase + i * 16 + hf * 8 + r;
      if (m < Mout) {
#pragma unroll
        for (int j = 0; j < 4; ++j) {
          int hw = hwb + j * 16 + mr;
          float vv = acc[i][j].e[r];
          if (outf) outf[((size_t)bp * Mout + m) * HW + hw] = vv;
          if (outh) outh[(size_t)(bp * HW + hw) * Cs + coff + m] = (_Float16)vv;
        }
      }
    }
}

// ---------------------------------------------------------------------------
// GroupNorm(32 groups of 8 ch over 64x64) + SiLU. One block per (b, group).
// Optional f32 NCHW output (may alias input) and f16 NHWC output.
// ---------------------------------------------------------------------------
__global__ __launch_bounds__(256)
void gn_silu(const float* __restrict__ in,
             const float* __restrict__ gamma,
             const float* __restrict__ beta,
             float* __restrict__ outf,
             _Float16* __restrict__ outh, int Cs, int coff) {
  __shared__ float sh[256], sh2[256];
  __shared__ float s_mean, s_rstd;
  const int tid = threadIdx.x;
  const int b = blockIdx.x >> 5, g = blockIdx.x & 31;
  const size_t base = ((size_t)b * CHN + g * 8) * HW;
  float s = 0.f, s2 = 0.f;
  for (int i = tid; i < 8 * HW; i += 256) {
    float v = in[base + i];
    s += v; s2 += v * v;
  }
  sh[tid] = s; sh2[tid] = s2;
  __syncthreads();
  for (int o = 128; o > 0; o >>= 1) {
    if (tid < o) { sh[tid] += sh[tid + o]; sh2[tid] += sh2[tid + o]; }
    __syncthreads();
  }
  if (tid == 0) {
    float mean = sh[0] / (8.0f * HW);
    float var  = sh2[0] / (8.0f * HW) - mean * mean;
    s_mean = mean;
    s_rstd = rsqrtf(var + 1e-6f);
  }
  __syncthreads();
  const float mean = s_mean, rstd = s_rstd;
  for (int i = tid; i < 8 * HW; i += 256) {
    int cl = i >> 12;
    int c = g * 8 + cl;
    int hw = i & (HW - 1);
    float v = in[base + i];
    float xn = (v - mean) * rstd * gamma[c] + beta[c];
    float ysi = xn / (1.0f + expf(-xn));
    if (outf) outf[base + i] = ysi;
    if (outh) {
      int n = b * HW + hw;
      outh[(size_t)n * Cs + coff + c] = (_Float16)ysi;
    }
  }
}

// Depthwise 7x7 pad 3. One block per (b,c).
__global__ __launch_bounds__(256)
void dwconv7(const float* __restrict__ in, const float* __restrict__ w,
             const float* __restrict__ bias, float* __restrict__ out) {
  __shared__ float wk[49];
  const int c = blockIdx.x & 255, b = blockIdx.x >> 8;
  if (threadIdx.x < 49) wk[threadIdx.x] = w[c * 49 + threadIdx.x];
  __syncthreads();
  const size_t base = ((size_t)b * CHN + c) * HW;
  const float bb = bias[c];
  for (int p = threadIdx.x; p < HW; p += 256) {
    int y = p >> 6, x = p & 63;
    float s = bb;
#pragma unroll
    for (int ky = 0; ky < 7; ++ky) {
      int yy = y + ky - 3;
      if (yy < 0 || yy >= 64) continue;
#pragma unroll
      for (int kx = 0; kx < 7; ++kx) {
        int xx = x + kx - 3;
        if (xx < 0 || xx >= 64) continue;
        s += wk[ky * 7 + kx] * in[base + yy * 64 + xx];
      }
    }
    out[base + p] = s;
  }
}

// Antialiased bilinear 1/2 downsample (jax.image.resize default): separable
// [1,3,3,1]/8 at input pos 2i+0.5, edge-renormalized. Output NHWC f16.
__global__ __launch_bounds__(256)
void resize_pack(const float* __restrict__ prior, _Float16* __restrict__ outh) {
  int idx = blockIdx.x * 256 + threadIdx.x;
  if (idx >= NPIX * 256) return;
  const int c = idx & 255, n = idx >> 8;
  const int b = n >> 12, hw = n & 4095, y = hw >> 6, x = hw & 63;
  const float* p = prior + (size_t)(b * 256 + c) * 128 * 128;
  const float W4[4] = {1.f, 3.f, 3.f, 1.f};
  float acc = 0.f, wsy = 0.f;
#pragma unroll
  for (int ty = 0; ty < 4; ++ty) {
    int yy = 2 * y - 1 + ty;
    if (yy < 0 || yy >= 128) continue;
    float ra = 0.f, wsx = 0.f;
#pragma unroll
    for (int tx = 0; tx < 4; ++tx) {
      int xx = 2 * x - 1 + tx;
      if (xx < 0 || xx >= 128) continue;
      ra += W4[tx] * p[yy * 128 + xx];
      wsx += W4[tx];
    }
    acc += W4[ty] * (ra / wsx);
    wsy += W4[ty];
  }
  outh[(size_t)n * 256 + c] = (_Float16)(acc / wsy);
}

// x_main (NCHW f32) -> concat buffer [NPIX][512] f16, channels 256..511
__global__ __launch_bounds__(256)
void pack_xmain(const float* __restrict__ x, _Float16* __restrict__ cath) {
  int idx = blockIdx.x * 256 + threadIdx.x;
  if (idx >= NPIX * 256) return;
  const int c = idx & 255, n = idx >> 8;
  const int b = n >> 12, hw = n & 4095;
  cath[(size_t)n * 512 + 256 + c] = (_Float16)x[((size_t)b * 256 + c) * HW + hw];
}

// plain f32 -> f16 convert
__global__ __launch_bounds__(256)
void conv_f16(const float* __restrict__ in, _Float16* __restrict__ out, int n) {
  int idx = blockIdx.x * 256 + threadIdx.x;
  if (idx < n) out[idx] = (_Float16)in[idx];
}

// (Mout,C,3,3) f32 -> [Mp][9][C] f16, zero rows for m>=Mout
__global__ __launch_bounds__(256)
void pack_w3(const float* __restrict__ w, _Float16* __restrict__ out,
             int Mp, int Mout, int C) {
  int idx = blockIdx.x * 256 + threadIdx.x;
  int total = Mp * 9 * C;
  if (idx >= total) return;
  int c = idx % C;
  int t = idx / C;
  int kp = t % 9;
  int m = t / 9;
  float v = (m < Mout) ? w[((size_t)m * C + c) * 9 + kp] : 0.0f;
  out[idx] = (_Float16)v;
}

// Deformable-conv bilinear gather: build val_t [NPIX][2304] f16 with
// K order (g*288 + c*9 + k) matching w_dcn's natural (C,3,3) flattening.
__global__ __launch_bounds__(256)
void dcn_gather(const float* __restrict__ x, const float* __restrict__ o,
                _Float16* __restrict__ valt) {
  int idx = blockIdx.x * 256 + threadIdx.x;
  if (idx >= 2 * 8 * 32 * HW) return;
  const int hw = idx & 4095;
  const int c  = (idx >> 12) & 31;
  const int g  = (idx >> 17) & 7;
  const int b  = idx >> 20;
  const int y = hw >> 6, xp = hw & 63;
  const float* xc = x + ((size_t)b * 256 + g * 32 + c) * HW;
  const float* ob = o + (size_t)b * 216 * HW;
  _Float16* vout = valt + ((size_t)(b * HW + hw)) * 2304 + g * 288 + c * 9;
#pragma unroll
  for (int k = 0; k < 9; ++k) {
    const int ky = k / 3 - 1, kx = k % 3 - 1;
    float offy = ob[(size_t)(g * 18 + 2 * k) * HW + hw];
    float offx = ob[(size_t)(g * 18 + 2 * k + 1) * HW + hw];
    float mm   = ob[(size_t)(144 + g * 9 + k) * HW + hw];
    float mask = 1.0f / (1.0f + expf(-mm));
    float py = (float)(y + ky) + offy;
    float px = (float)(xp + kx) + offx;
    float y0f = floorf(py), x0f = floorf(px);
    float wy1 = py - y0f, wx1 = px - x0f;
    int y0 = (int)y0f, x0 = (int)x0f;
    float v = 0.f;
    auto tap = [&](int yi, int xi, float wt) {
      if (yi >= 0 && yi < 64 && xi >= 0 && xi < 64) v += wt * xc[yi * 64 + xi];
    };
    tap(y0,     x0,     (1.f - wy1) * (1.f - wx1));
    tap(y0,     x0 + 1, (1.f - wy1) * wx1);
    tap(y0 + 1, x0,     wy1 * (1.f - wx1));
    tap(y0 + 1, x0 + 1, wy1 * wx1);
    vout[k] = (_Float16)(v * mask);
  }
}

// ---------------------------------------------------------------------------
extern "C" void kernel_launch(void* const* d_in, const int* in_sizes, int n_in,
                              void* d_out, int out_size, void* d_ws, size_t ws_size,
                              hipStream_t stream) {
  const float* x_main = (const float*)d_in[0];
  const float* prior  = (const float*)d_in[1];
  const float* w_ds   = (const float*)d_in[2];
  const float* b_ds   = (const float*)d_in[3];
  const float* w1     = (const float*)d_in[4];
  const float* b1     = (const float*)d_in[5];
  const float* g1     = (const float*)d_in[6];
  const float* bt1    = (const float*)d_in[7];
  const float* w_dw   = (const float*)d_in[8];
  const float* b_dw   = (const float*)d_in[9];
  const float* g2     = (const float*)d_in[10];
  const float* bt2    = (const float*)d_in[11];
  const float* w2     = (const float*)d_in[12];
  const float* b2     = (const float*)d_in[13];
  const float* w3     = (const float*)d_in[14];
  const float* b3     = (const float*)d_in[15];
  const float* g3     = (const float*)d_in[16];
  const float* bt3    = (const float*)d_in[17];
  const float* w_off  = (const float*)d_in[18];
  const float* b_off  = (const float*)d_in[19];
  const float* w_dcn  = (const float*)d_in[20];
  const float* b_dcn  = (const float*)d_in[21];

  float* out_warp = (float*)d_out;
  float* out_offf = (float*)d_out + (size_t)2 * CHN * HW;

  char* wsp = (char*)d_ws;
  auto alloc = [&](size_t bytes) {
    void* p = (void*)wsp;
    wsp += (bytes + 255) & ~(size_t)255;
    return p;
  };
  _Float16* pr0h  = (_Float16*)alloc((size_t)NPIX * 256 * 2);   // resized prior, NHWC f16
  _Float16* cath  = (_Float16*)alloc((size_t)NPIX * 512 * 2);   // [pr ; x_main] NHWC f16
  float*    tf    = (float*)   alloc((size_t)NPIX * 256 * 4);   // t, NCHW f32
  float*    t2f   = (float*)   alloc((size_t)NPIX * 256 * 4);   // dw / w3 out, NCHW f32
  _Float16* th    = (_Float16*)alloc((size_t)NPIX * 256 * 2);   // NHWC f16 staging A
  _Float16* th2   = (_Float16*)alloc((size_t)NPIX * 256 * 2);   // NHWC f16 staging B
  float*    of    = (float*)   alloc((size_t)NPIX * 216 * 4);   // offsets/mask, NCHW f32
  _Float16* valt  = (_Float16*)alloc((size_t)NPIX * 2304 * 2);  // DCN gathered vals
  _Float16* wdsp  = (_Float16*)alloc((size_t)256 * 9 * 256 * 2);
  _Float16* w1h   = (_Float16*)alloc((size_t)256 * 512 * 2);
  _Float16* w2h   = (_Float16*)alloc((size_t)256 * 256 * 2);
  _Float16* w3p   = (_Float16*)alloc((size_t)256 * 9 * 256 * 2);
  _Float16* woffp = (_Float16*)alloc((size_t)224 * 9 * 256 * 2);
  _Float16* wdcnh = (_Float16*)alloc((size_t)256 * 2304 * 2);

  const dim3 blk(256);
  const dim3 gGemm(16, 8);    // (8192/512) x (256/32)
  const dim3 gOff(16, 7);     // 224/32 M-tiles for w_off

  // 1) antialiased bilinear downsample of prior -> NHWC f16
  resize_pack<<<NPIX * 256 / 256, blk, 0, stream>>>(prior, pr0h);
  // 2) 3x3 conv on pr (256->256), result into concat buffer channels [0,256)
  pack_w3<<<(256 * 9 * 256 + 255) / 256, blk, 0, stream>>>(w_ds, wdsp, 256, 256, 256);
  conv3x3_wmma<<<gGemm, blk, 0, stream>>>(wdsp, pr0h, b_ds, nullptr, cath, 512, 0, 256, 256);
  // 3) x_main into concat channels [256,512)
  pack_xmain<<<NPIX * 256 / 256, blk, 0, stream>>>(x_main, cath);
  // 4) 1x1 conv (512->256): t
  conv_f16<<<(256 * 512 + 255) / 256, blk, 0, stream>>>(w1, w1h, 256 * 512);
  gemm_nt_wmma<<<gGemm, blk, 0, stream>>>(w1h, cath, b1, tf, nullptr, 1, 0, 256, 512);
  // 5) GN1 + SiLU (in place, f32)
  gn_silu<<<64, blk, 0, stream>>>(tf, g1, bt1, tf, nullptr, 1, 0);
  // 6) depthwise 7x7
  dwconv7<<<512, blk, 0, stream>>>(tf, w_dw, b_dw, t2f);
  // 7) GN2 + SiLU -> NHWC f16 (feeds w2 GEMM)
  gn_silu<<<64, blk, 0, stream>>>(t2f, g2, bt2, nullptr, th, 256, 0);
  // 8) 1x1 conv (256->256) -> NHWC f16 (feeds w3 conv)
  conv_f16<<<(256 * 256 + 255) / 256, blk, 0, stream>>>(w2, w2h, 256 * 256);
  gemm_nt_wmma<<<gGemm, blk, 0, stream>>>(w2h, th, b2, nullptr, th2, 256, 0, 256, 256);
  // 9) 3x3 conv (256->256) -> f32
  pack_w3<<<(256 * 9 * 256 + 255) / 256, blk, 0, stream>>>(w3, w3p, 256, 256, 256);
  conv3x3_wmma<<<gGemm, blk, 0, stream>>>(w3p, th2, b3, t2f, nullptr, 1, 0, 256, 256);
  // 10) GN3 + SiLU -> offset_feat (output) + NHWC f16 (feeds w_off conv)
  gn_silu<<<64, blk, 0, stream>>>(t2f, g3, bt3, out_offf, th, 256, 0);
  // 11) 3x3 conv (256->216): offsets + mask logits
  pack_w3<<<(224 * 9 * 256 + 255) / 256, blk, 0, stream>>>(w_off, woffp, 224, 216, 256);
  conv3x3_wmma<<<gOff, blk, 0, stream>>>(woffp, th, b_off, of, nullptr, 1, 0, 216, 256);
  // 12) deformable bilinear gather -> val_t [NPIX][2304] f16
  dcn_gather<<<2 * 8 * 32 * HW / 256, blk, 0, stream>>>(x_main, of, valt);
  // 13) DCN einsum as GEMM (K=2304) -> warp_feat (output)
  conv_f16<<<(256 * 2304 + 255) / 256, blk, 0, stream>>>(w_dcn, wdcnh, 256 * 2304);
  gemm_nt_wmma<<<gGemm, blk, 0, stream>>>(wdcnh, valt, b_dcn, out_warp, nullptr, 1, 0, 256, 2304);
}
